// myWholeRGAT_43877385896326
// MI455X (gfx1250) — matrix operand order repaired and checked by reading, not compiled, not run on hardware
//
#include <hip/hip_runtime.h>

// ---------------------------------------------------------------------------
// RGAT (3 layers) on dense bipartite graphs, B=4, N=256/graph, F=256.
// All dense contractions run on v_wmma_f32_16x16x32_f16 (wave32 WMMA).
// ---------------------------------------------------------------------------

typedef _Float16 h8_t   __attribute__((ext_vector_type(8)));
typedef _Float16 h16_t  __attribute__((ext_vector_type(16)));
typedef float    f8_t   __attribute__((ext_vector_type(8)));

#define SHUF16(lo, hi) __builtin_shufflevector(lo, hi, 0,1,2,3,4,5,6,7,8,9,10,11,12,13,14,15)

// A-fragment (16x32, row-major source [M,K]):
//  lanes 0-15 hold row M=lane:   elems 0..7 = K k0+0..7,  8..15 = K k0+16..23
//  lanes 16-31 hold row M=lane-16: elems 0..7 = K k0+8..15, 8..15 = K k0+24..31
__device__ __forceinline__ h16_t load_a_frag(const _Float16* __restrict__ A, int lda,
                                             int row, int k0, int hi) {
  const _Float16* p = A + (size_t)row * lda + k0 + hi * 8;
  h8_t lo = *(const h8_t*)(p);
  h8_t h2 = *(const h8_t*)(p + 16);
  return SHUF16(lo, h2);
}

// B-fragment (32x16) from Bt stored as [N][K] (i.e. B transposed):
//  lanes 0-15: column N=lane, elems 0..7 = K k0+0..7, 8..15 = K k0+16..23; lanes 16-31: K k0+8..15, k0+24..31
__device__ __forceinline__ h16_t load_b_frag(const _Float16* __restrict__ Bt, int ldb,
                                             int col, int k0, int hi) {
  const _Float16* p = Bt + (size_t)col * ldb + k0 + hi * 8;
  h8_t lo = *(const h8_t*)(p);
  h8_t h2 = *(const h8_t*)(p + 16);
  return SHUF16(lo, h2);
}

__device__ __forceinline__ f8_t wmma_f32_16x16x32(h16_t a, h16_t b, f8_t c) {
  return __builtin_amdgcn_wmma_f32_16x16x32_f16(false, a, false, b, (short)0, c, false, false);
}

// ---------------------------------------------------------------------------
// Pack desc0/desc1 [B,F,128] -> x [1024,256] f32 and cat[:, :256] f16
// ---------------------------------------------------------------------------
__global__ __launch_bounds__(256) void k_pack(const float* __restrict__ d0,
                                              const float* __restrict__ d1,
                                              float* __restrict__ x,
                                              _Float16* __restrict__ cat) {
  int idx  = blockIdx.x * 256 + threadIdx.x;     // 0..262143
  int node = idx >> 8, f = idx & 255;
  int b = node >> 8, i = node & 255;
  float v = (i < 128) ? d0[b * 32768 + f * 128 + i]
                      : d1[b * 32768 + f * 128 + (i - 128)];
  x[idx] = v;
  cat[(size_t)node * 512 + f] = (_Float16)v;
}

// conv_w [G=6][K=256][N=256] f32 -> Wt [G][N][K] f16 (transposed per matrix)
__global__ __launch_bounds__(256) void k_convert_wt(const float* __restrict__ src,
                                                    _Float16* __restrict__ dst) {
  int idx = blockIdx.x * 256 + threadIdx.x;      // 0..393215
  int g = idx >> 16, rem = idx & 65535;
  int n = rem >> 8, k = rem & 255;
  dst[idx] = (_Float16)src[g * 65536 + k * 256 + n];
}

// straight f32 -> f16 convert (lin_w is already [out][in] = Bt layout)
__global__ __launch_bounds__(256) void k_convert(const float* __restrict__ src,
                                                 _Float16* __restrict__ dst, int n) {
  int idx = blockIdx.x * 256 + threadIdx.x;
  if (idx < n) dst[idx] = (_Float16)src[idx];
}

// ---------------------------------------------------------------------------
// Transform GEMM: H_r = x @ W_r  (M=1024 nodes, K=256, N=256 feats, z=rel)
// D stored transposed as f16 into Ht[f][b][rel][s]  (== B operand for agg)
// ---------------------------------------------------------------------------
__global__ __launch_bounds__(256) void k_gemm_transform(const _Float16* __restrict__ cat,
                                                        const _Float16* __restrict__ Wt,
                                                        _Float16* __restrict__ Ht) {
  const int lane = threadIdx.x & 31, wave = threadIdx.x >> 5;
  const int hi = lane >> 4, lm = lane & 15;
  const int mBase = blockIdx.x * 128 + wave * 16;
  const int nBase = blockIdx.y * 64;
  const int rel = blockIdx.z;
  const _Float16* Bt = Wt + rel * 65536;

  f8_t c0 = {}, c1 = {}, c2 = {}, c3 = {};
  for (int k0 = 0; k0 < 256; k0 += 32) {
    h16_t a  = load_a_frag(cat, 512, mBase + lm, k0, hi);
    h16_t b0 = load_b_frag(Bt, 256, nBase + 0  + lm, k0, hi);
    h16_t b1 = load_b_frag(Bt, 256, nBase + 16 + lm, k0, hi);
    h16_t b2 = load_b_frag(Bt, 256, nBase + 32 + lm, k0, hi);
    h16_t b3 = load_b_frag(Bt, 256, nBase + 48 + lm, k0, hi);
    c0 = wmma_f32_16x16x32(a, b0, c0);
    c1 = wmma_f32_16x16x32(a, b1, c1);
    c2 = wmma_f32_16x16x32(a, b2, c2);
    c3 = wmma_f32_16x16x32(a, b3, c3);
  }
  f8_t acc[4] = {c0, c1, c2, c3};
#pragma unroll
  for (int t = 0; t < 4; ++t) {
    int n = nBase + t * 16 + lm;
#pragma unroll
    for (int rr = 0; rr < 8; ++rr) {
      int m = mBase + hi * 8 + rr;                     // global node
      Ht[(size_t)n * 2048 + (m >> 8) * 512 + rel * 256 + (m & 255)] = (_Float16)acc[t][rr];
    }
  }
}

// ---------------------------------------------------------------------------
// qs[rel][node] = H_rel[node] . q ; ks analogously   (reads transposed Ht)
// ---------------------------------------------------------------------------
__global__ __launch_bounds__(256) void k_qk(const _Float16* __restrict__ Ht,
                                            const float* __restrict__ q,
                                            const float* __restrict__ k,
                                            float* __restrict__ qs,
                                            float* __restrict__ ks) {
  int idx  = blockIdx.x * 256 + threadIdx.x;     // 0..2047
  int rel  = idx >> 10, node = idx & 1023;
  int b = node >> 8, s = node & 255;
  const _Float16* p = Ht + b * 512 + rel * 256 + s;
  float aq = 0.f, ak = 0.f;
  for (int f = 0; f < 256; ++f) {
    float h = (float)p[(size_t)f * 2048];
    aq += h * q[f];
    ak += h * k[f];
  }
  qs[idx] = aq;
  ks[idx] = ak;
}

// ---------------------------------------------------------------------------
// Per-dst softmax over the 255 in-edges (union over both relations),
// writing dense masked Alpha[b][j][rel*256+s] in f16 (zeros = no edge).
// ---------------------------------------------------------------------------
__global__ __launch_bounds__(256) void k_alpha(const float* __restrict__ qs,
                                               const float* __restrict__ ks,
                                               _Float16* __restrict__ Al) {
  int bj = blockIdx.x;                 // 0..1023
  int b = bj >> 8, j = bj & 255;
  int s = threadIdx.x;                 // src 0..255
  bool sideJ = j >= 128, sideS = s >= 128;
  int nodeJ = b * 256 + j, nodeS = b * 256 + s;
  float q0 = qs[nodeJ], q1 = qs[1024 + nodeJ];
  float k0 = ks[nodeS], k1 = ks[1024 + nodeS];
  bool v0 = (sideS == sideJ) && (s != j);   // intra edge (rel 0)
  bool v1 = (sideS != sideJ);               // cross edge (rel 1)
  float t0 = q0 + k0, t1 = q1 + k1;
  float l0 = t0 > 0.f ? t0 : 0.2f * t0;     // leaky_relu 0.2
  float l1 = t1 > 0.f ? t1 : 0.2f * t1;

  __shared__ float red[256];
  red[s] = fmaxf(v0 ? l0 : -1e30f, v1 ? l1 : -1e30f);
  __syncthreads();
  for (int off = 128; off; off >>= 1) {
    if (s < off) red[s] = fmaxf(red[s], red[s + off]);
    __syncthreads();
  }
  float M = red[0];
  __syncthreads();

  float e0 = v0 ? __expf(l0 - M) : 0.f;
  float e1 = v1 ? __expf(l1 - M) : 0.f;
  red[s] = e0 + e1;
  __syncthreads();
  for (int off = 128; off; off >>= 1) {
    if (s < off) red[s] += red[s + off];
    __syncthreads();
  }
  float Z = red[0] + 1e-16f;

  size_t base = (size_t)(b * 256 + j) * 512;
  Al[base + s]       = (_Float16)(e0 / Z);
  Al[base + 256 + s] = (_Float16)(e1 / Z);
}

// ---------------------------------------------------------------------------
// Aggregation GEMM per graph: out = [Alpha0|Alpha1] @ [H0;H1]  (M=256,K=512,N=256)
// Epilogue: relu(out + conv_b) -> f16 into cat[:, 256:512]
// ---------------------------------------------------------------------------
__global__ __launch_bounds__(256) void k_gemm_agg(const _Float16* __restrict__ Al,
                                                  const _Float16* __restrict__ Ht,
                                                  const float* __restrict__ bias,
                                                  _Float16* __restrict__ cat) {
  const int lane = threadIdx.x & 31, wave = threadIdx.x >> 5;
  const int hi = lane >> 4, lm = lane & 15;
  const int mBase = blockIdx.x * 128 + wave * 16;       // 0..255 (local node)
  const int nBase = blockIdx.y * 64;
  const int b = blockIdx.z;
  const _Float16* Ablk = Al + (size_t)b * 131072;       // [256][512]
  const _Float16* Bt   = Ht + b * 512;                  // row n: Ht[n*2048 + b*512 + k]

  f8_t c0 = {}, c1 = {}, c2 = {}, c3 = {};
  for (int k0 = 0; k0 < 512; k0 += 32) {
    h16_t a  = load_a_frag(Ablk, 512, mBase + lm, k0, hi);
    h16_t b0 = load_b_frag(Bt, 2048, nBase + 0  + lm, k0, hi);
    h16_t b1 = load_b_frag(Bt, 2048, nBase + 16 + lm, k0, hi);
    h16_t b2 = load_b_frag(Bt, 2048, nBase + 32 + lm, k0, hi);
    h16_t b3 = load_b_frag(Bt, 2048, nBase + 48 + lm, k0, hi);
    c0 = wmma_f32_16x16x32(a, b0, c0);
    c1 = wmma_f32_16x16x32(a, b1, c1);
    c2 = wmma_f32_16x16x32(a, b2, c2);
    c3 = wmma_f32_16x16x32(a, b3, c3);
  }
  f8_t acc[4] = {c0, c1, c2, c3};
#pragma unroll
  for (int t = 0; t < 4; ++t) {
    int n = nBase + t * 16 + lm;
    float bn = bias[n];
#pragma unroll
    for (int rr = 0; rr < 8; ++rr) {
      int m = mBase + hi * 8 + rr;
      float v = acc[t][rr] + bn;
      v = fmaxf(v, 0.f);                                // relu(msg1)
      cat[(size_t)(b * 256 + m) * 512 + 256 + n] = (_Float16)v;
    }
  }
}

// ---------------------------------------------------------------------------
// Linear GEMM: msg2 = cat @ lin_w^T + lin_b   (M=1024, K=512, N=256)
// ---------------------------------------------------------------------------
__global__ __launch_bounds__(256) void k_gemm_lin(const _Float16* __restrict__ cat,
                                                  const _Float16* __restrict__ Bt,
                                                  const float* __restrict__ bias,
                                                  float* __restrict__ msg2) {
  const int lane = threadIdx.x & 31, wave = threadIdx.x >> 5;
  const int hi = lane >> 4, lm = lane & 15;
  const int mBase = blockIdx.x * 128 + wave * 16;
  const int nBase = blockIdx.y * 64;

  f8_t c0 = {}, c1 = {}, c2 = {}, c3 = {};
  for (int k0 = 0; k0 < 512; k0 += 32) {
    h16_t a  = load_a_frag(cat, 512, mBase + lm, k0, hi);
    h16_t b0 = load_b_frag(Bt, 512, nBase + 0  + lm, k0, hi);
    h16_t b1 = load_b_frag(Bt, 512, nBase + 16 + lm, k0, hi);
    h16_t b2 = load_b_frag(Bt, 512, nBase + 32 + lm, k0, hi);
    h16_t b3 = load_b_frag(Bt, 512, nBase + 48 + lm, k0, hi);
    c0 = wmma_f32_16x16x32(a, b0, c0);
    c1 = wmma_f32_16x16x32(a, b1, c1);
    c2 = wmma_f32_16x16x32(a, b2, c2);
    c3 = wmma_f32_16x16x32(a, b3, c3);
  }
  f8_t acc[4] = {c0, c1, c2, c3};
#pragma unroll
  for (int t = 0; t < 4; ++t) {
    int n = nBase + t * 16 + lm;
    float bn = bias[n];
#pragma unroll
    for (int rr = 0; rr < 8; ++rr) {
      int m = mBase + hi * 8 + rr;
      msg2[(size_t)m * 256 + n] = acc[t][rr] + bn;
    }
  }
}

// ---------------------------------------------------------------------------
// BatchNorm (training-mode, biased var) column stats over 1024 rows
// ---------------------------------------------------------------------------
__global__ __launch_bounds__(256) void k_bn_stats(const float* __restrict__ msg2,
                                                  float* __restrict__ mu,
                                                  float* __restrict__ rstd) {
  int f = blockIdx.x, t = threadIdx.x;
  float s = 0.f, s2 = 0.f;
  for (int r = t; r < 1024; r += 256) {
    float v = msg2[(size_t)r * 256 + f];
    s += v; s2 += v * v;
  }
  __shared__ float rs[256], rq[256];
  rs[t] = s; rq[t] = s2;
  __syncthreads();
  for (int off = 128; off; off >>= 1) {
    if (t < off) { rs[t] += rs[t + off]; rq[t] += rq[t + off]; }
    __syncthreads();
  }
  if (t == 0) {
    float m = rs[0] * (1.f / 1024.f);
    float var = rq[0] * (1.f / 1024.f) - m * m;
    mu[f] = m;
    rstd[f] = rsqrtf(var + 1e-5f);
  }
}

// x += bn_w*(msg2-mu)*rstd + bn_b ; refresh f16 copy in cat[:, :256]
__global__ __launch_bounds__(256) void k_bn_apply(float* __restrict__ x,
                                                  const float* __restrict__ msg2,
                                                  const float* __restrict__ mu,
                                                  const float* __restrict__ rstd,
                                                  const float* __restrict__ g,
                                                  const float* __restrict__ be,
                                                  _Float16* __restrict__ cat) {
  int idx = blockIdx.x * 256 + threadIdx.x;      // 0..262143
  int node = idx >> 8, f = idx & 255;
  float v = x[idx] + g[f] * (msg2[idx] - mu[f]) * rstd[f] + be[f];
  x[idx] = v;
  cat[(size_t)node * 512 + f] = (_Float16)v;
}

// x [1024,256] -> out (= desc0' [4,256,128] ++ desc1' [4,256,128])
__global__ __launch_bounds__(256) void k_unpack(const float* __restrict__ x,
                                                float* __restrict__ out) {
  int idx = blockIdx.x * 256 + threadIdx.x;      // 0..262143
  int half = idx >= 131072 ? 1 : 0;
  int r = idx - half * 131072;
  int b = r >> 15, rem = r & 32767, f = rem >> 7, i = rem & 127;
  out[idx] = x[(size_t)(b * 256 + half * 128 + i) * 256 + f];
}

// ---------------------------------------------------------------------------
extern "C" void kernel_launch(void* const* d_in, const int* in_sizes, int n_in,
                              void* d_out, int out_size, void* d_ws, size_t ws_size,
                              hipStream_t stream) {
  const float* desc0  = (const float*)d_in[0];
  const float* desc1  = (const float*)d_in[1];
  const float* conv_w = (const float*)d_in[2];
  const float* conv_q = (const float*)d_in[3];
  const float* conv_k = (const float*)d_in[4];
  const float* conv_b = (const float*)d_in[5];
  const float* lin_w  = (const float*)d_in[6];
  const float* lin_b  = (const float*)d_in[7];
  const float* bn_w   = (const float*)d_in[8];
  const float* bn_b   = (const float*)d_in[9];
  float* out = (float*)d_out;

  char* ws = (char*)d_ws;
  float*    x     = (float*)   (ws + 0);         // 1 MB  [1024][256] f32
  _Float16* cat   = (_Float16*)(ws + 1048576);   // 1 MB  [1024][512] f16 (x | msg1)
  _Float16* Wt    = (_Float16*)(ws + 2097152);   // 768KB [3][2][256n][256k] f16
  _Float16* linWh = (_Float16*)(ws + 2883584);   // 768KB [3][256][512] f16
  _Float16* Ht    = (_Float16*)(ws + 3670016);   // 1 MB  [256f][4b][2r][256s] f16
  float*    qs    = (float*)   (ws + 4718592);   // 8 KB  [2][1024]
  float*    ks    = (float*)   (ws + 4726784);   // 8 KB  [2][1024]
  _Float16* Al    = (_Float16*)(ws + 4734976);   // 1 MB  [4][256][512] f16
  float*    msg2  = (float*)   (ws + 5783552);   // 1 MB  [1024][256]
  float*    mu    = (float*)   (ws + 6832128);   // 1 KB
  float*    rstd  = (float*)   (ws + 6833152);   // 1 KB

  k_pack<<<1024, 256, 0, stream>>>(desc0, desc1, x, cat);
  k_convert_wt<<<1536, 256, 0, stream>>>(conv_w, Wt);          // 3*2*256*256
  k_convert<<<1536, 256, 0, stream>>>(lin_w, linWh, 393216);   // 3*256*512

  for (int l = 0; l < 3; ++l) {
    k_gemm_transform<<<dim3(8, 4, 2), 256, 0, stream>>>(cat, Wt + (size_t)l * 131072, Ht);
    k_qk<<<8, 256, 0, stream>>>(Ht, conv_q + l * 256, conv_k + l * 256, qs, ks);
    k_alpha<<<1024, 256, 0, stream>>>(qs, ks, Al);
    k_gemm_agg<<<dim3(2, 4, 4), 256, 0, stream>>>(Al, Ht, conv_b + l * 256, cat);
    k_gemm_lin<<<dim3(8, 4, 1), 256, 0, stream>>>(cat, linWh + (size_t)l * 131072,
                                                  lin_b + l * 256, msg2);
    k_bn_stats<<<256, 256, 0, stream>>>(msg2, mu, rstd);
    k_bn_apply<<<1024, 256, 0, stream>>>(x, msg2, mu, rstd, bn_w + l * 256,
                                         bn_b + l * 256, cat);
  }

  k_unpack<<<1024, 256, 0, stream>>>(x, out);
}
